// InPlaceTTT_MLP_69870527972018
// MI455X (gfx1250) — matrix operand order repaired and checked by reading, not compiled, not run on hardware
//
#include <hip/hip_runtime.h>
#include <stdint.h>

// ---------------- problem constants ----------------
constexpr int Bc = 2;
constexpr int Sc = 2048;
constexpr int Hc = 2048;
constexpr int Ic = 8192;
constexpr int Rc = 16;
constexpr int Mc = Bc * Sc;          // 4096 rows
constexpr float TTT_LR = 0.01f;

// ---------------- feature detection ----------------
#if defined(__has_builtin)
#if __has_builtin(__builtin_amdgcn_global_load_async_to_lds_b128)
#define HAVE_ASYNC_LDS 1
#endif
#endif
#ifndef HAVE_ASYNC_LDS
#define HAVE_ASYNC_LDS 0
#endif

// ---------------- types ----------------
typedef __bf16 v16bf __attribute__((ext_vector_type(16)));
typedef float  v8f   __attribute__((ext_vector_type(8)));
typedef float  vf4   __attribute__((ext_vector_type(4)));
typedef int    v4i   __attribute__((ext_vector_type(4)));

#if HAVE_ASYNC_LDS
typedef __attribute__((address_space(1))) v4i* gv4ip;   // global int4*
typedef __attribute__((address_space(3))) v4i* lv4ip;   // LDS int4*
#endif

struct alignas(16) F4 { float x, y, z, w; };
struct alignas(16) F8 { F4 lo, hi; };
struct alignas(8)  BF4 { __bf16 v[4]; };

static_assert(sizeof(F8) == sizeof(v16bf), "frag size");

__device__ inline v16bf to_v16bf(F4 lo, F4 hi) {
  F8 t{lo, hi};
  return __builtin_bit_cast(v16bf, t);
}

// A-matrix 16x32 bf16 fragment (ISA 7.12.2): lanes 0-15 row m hold K{0..7,16..23},
// lanes 16-31 hold K{8..15,24..31}.
__device__ inline v16bf load_fragA(const __bf16* row, int half) {
  F4 lo = *(const F4*)(row + half * 8);
  F4 hi = *(const F4*)(row + 16 + half * 8);
  return to_v16bf(lo, hi);
}
// B-matrix 32x16 bf16 fragment: lane group 0-15 holds K{0..7,16..23}, 16-31 holds K{8..15,24..31}
// (column n = lane&15, contiguous K in memory since B given as [N,K] row-major).
__device__ inline v16bf load_fragB(const __bf16* row, int half) {
  F4 lo = *(const F4*)(row + half * 8);
  F4 hi = *(const F4*)(row + 16 + half * 8);
  return to_v16bf(lo, hi);
}

__device__ inline v8f wmma_bf16(v16bf a, v16bf b, v8f c) {
  return __builtin_amdgcn_wmma_f32_16x16x32_bf16(false, a, false, b, (short)0, c,
                                                 false, false);
}

#if HAVE_ASYNC_LDS
// 16B per-lane async copy global -> LDS, tracked by ASYNCcnt.
// Signature (per toolchain diagnostic): (v4i as1*, v4i as3*, imm offset, imm cpol).
__device__ inline void async_cp16(const __bf16* g, __bf16* l) {
  __builtin_amdgcn_global_load_async_to_lds_b128(
      (gv4ip)(uintptr_t)g,     // global generic addr == AS1 addr
      (lv4ip)(uintptr_t)l,     // low 32 bits of generic shared addr == LDS offset
      0, 0);
}
__device__ inline void wait_async0() {
#if __has_builtin(__builtin_amdgcn_s_wait_asynccnt)
  __builtin_amdgcn_s_wait_asynccnt(0);
#else
  asm volatile("s_wait_asynccnt 0x0" ::: "memory");
#endif
}
#endif

// ---------------- fp32 -> bf16 convert (NT streaming reads) ----------------
__global__ __launch_bounds__(256)
void cvt_bf16_kernel(const float* __restrict__ s, __bf16* __restrict__ d, int n4) {
  int i = blockIdx.x * 256 + threadIdx.x;
  if (i < n4) {
    vf4 f = __builtin_nontemporal_load((const vf4*)(s + (size_t)i * 4));
    BF4 o;
    o.v[0] = (__bf16)f.x; o.v[1] = (__bf16)f.y;
    o.v[2] = (__bf16)f.z; o.v[3] = (__bf16)f.w;
    *(BF4*)(d + (size_t)i * 4) = o;
  }
}

// ---------------- dual-B tiled WMMA GEMM ----------------
// C1[m,n] = sum_k A[m,k]*B1[n,k] ; C2 likewise with B2 (A,B bf16 row-major, K contig).
// MODE 0: epilogue z = silu(C1)*C2 -> bf16 out1bf
// MODE 1: epilogue C1 -> out1f (fp32), C2 -> out2f (fp32)
template <int MODE>
__global__ __launch_bounds__(256)
void gemm_dual_kernel(const __bf16* __restrict__ A,
                      const __bf16* __restrict__ B1,
                      const __bf16* __restrict__ B2,
                      int K, int N,
                      float* __restrict__ out1f, float* __restrict__ out2f,
                      __bf16* __restrict__ out1bf) {
  constexpr int BM = 64, BN = 128, PAD = 40;   // PAD=40 elts => conflict-free b128 reads
  __shared__ __bf16 As [2][BM * PAD];
  __shared__ __bf16 Bs1[2][BN * PAD];
  __shared__ __bf16 Bs2[2][BN * PAD];

  const int tid  = threadIdx.x;
  const int w    = tid >> 5, l = tid & 31;
  const int wm   = w & 1,    wn = w >> 1;      // 2 x 4 wave grid
  const int half = l >> 4,   lm = l & 15;
  const size_t m0 = (size_t)blockIdx.y * BM;
  const size_t n0 = (size_t)blockIdx.x * BN;

  // staging: each thread moves one 16B chunk of A and two of each B per K-step
  const int srow = tid >> 2;            // 0..63
  const int scol = (tid & 3) * 8;       // 0,8,16,24
  const int sidx = srow * PAD + scol;

  const __bf16* gA   = A  + (m0 + srow)      * (size_t)K + scol;
  const __bf16* gB1a = B1 + (n0 + srow)      * (size_t)K + scol;
  const __bf16* gB1b = B1 + (n0 + 64 + srow) * (size_t)K + scol;
  const __bf16* gB2a = B2 + (n0 + srow)      * (size_t)K + scol;
  const __bf16* gB2b = B2 + (n0 + 64 + srow) * (size_t)K + scol;

  v8f acc1[2][2], acc2[2][2];
#pragma unroll
  for (int i = 0; i < 2; ++i)
#pragma unroll
    for (int j = 0; j < 2; ++j)
#pragma unroll
      for (int e = 0; e < 8; ++e) { acc1[i][j][e] = 0.f; acc2[i][j][e] = 0.f; }

  const int KT = K >> 5;

#if HAVE_ASYNC_LDS
  // -------- async copy pipeline: global -> LDS with ASYNCcnt tracking --------
  auto stage = [&](int bufi, size_t ko) {
    async_cp16(gA   + ko, &As [bufi][sidx]);
    async_cp16(gB1a + ko, &Bs1[bufi][sidx]);
    async_cp16(gB1b + ko, &Bs1[bufi][sidx + 64 * PAD]);
    async_cp16(gB2a + ko, &Bs2[bufi][sidx]);
    async_cp16(gB2b + ko, &Bs2[bufi][sidx + 64 * PAD]);
  };
  stage(0, 0);
  wait_async0();
  __syncthreads();

  for (int kt = 0; kt < KT; ++kt) {
    const int cur = kt & 1;
    if (kt + 1 < KT) stage(cur ^ 1, (size_t)(kt + 1) * 32);  // overlap with WMMA

    v16bf af[2];
#pragma unroll
    for (int i = 0; i < 2; ++i)
      af[i] = load_fragA(&As[cur][(wm * 32 + i * 16 + lm) * PAD], half);
#pragma unroll
    for (int j = 0; j < 2; ++j) {
      v16bf b1 = load_fragB(&Bs1[cur][(wn * 32 + j * 16 + lm) * PAD], half);
      v16bf b2 = load_fragB(&Bs2[cur][(wn * 32 + j * 16 + lm) * PAD], half);
#pragma unroll
      for (int i = 0; i < 2; ++i) {
        acc1[i][j] = wmma_bf16(af[i], b1, acc1[i][j]);
        acc2[i][j] = wmma_bf16(af[i], b2, acc2[i][j]);
      }
    }
    wait_async0();
    __syncthreads();
  }
#else
  // -------- fallback: register-prefetched manual staging --------
  {
    *(F4*)&As [0][sidx]            = *(const F4*)gA;
    *(F4*)&Bs1[0][sidx]            = *(const F4*)gB1a;
    *(F4*)&Bs1[0][sidx + 64 * PAD] = *(const F4*)gB1b;
    *(F4*)&Bs2[0][sidx]            = *(const F4*)gB2a;
    *(F4*)&Bs2[0][sidx + 64 * PAD] = *(const F4*)gB2b;
  }
  __syncthreads();

  for (int kt = 0; kt < KT; ++kt) {
    const int cur = kt & 1;
    const bool pf = (kt + 1 < KT);
    F4 ra, r1a, r1b, r2a, r2b;
    if (pf) {                                    // issue next-tile loads early
      const size_t ko = (size_t)(kt + 1) * 32;
      ra  = *(const F4*)(gA   + ko);
      r1a = *(const F4*)(gB1a + ko);
      r1b = *(const F4*)(gB1b + ko);
      r2a = *(const F4*)(gB2a + ko);
      r2b = *(const F4*)(gB2b + ko);
    }

    v16bf af[2];
#pragma unroll
    for (int i = 0; i < 2; ++i)
      af[i] = load_fragA(&As[cur][(wm * 32 + i * 16 + lm) * PAD], half);
#pragma unroll
    for (int j = 0; j < 2; ++j) {
      v16bf b1 = load_fragB(&Bs1[cur][(wn * 32 + j * 16 + lm) * PAD], half);
      v16bf b2 = load_fragB(&Bs2[cur][(wn * 32 + j * 16 + lm) * PAD], half);
#pragma unroll
      for (int i = 0; i < 2; ++i) {
        acc1[i][j] = wmma_bf16(af[i], b1, acc1[i][j]);
        acc2[i][j] = wmma_bf16(af[i], b2, acc2[i][j]);
      }
    }

    if (pf) {                                    // commit into other buffer
      const int nb = cur ^ 1;
      *(F4*)&As [nb][sidx]            = ra;
      *(F4*)&Bs1[nb][sidx]            = r1a;
      *(F4*)&Bs1[nb][sidx + 64 * PAD] = r1b;
      *(F4*)&Bs2[nb][sidx]            = r2a;
      *(F4*)&Bs2[nb][sidx + 64 * PAD] = r2b;
    }
    __syncthreads();
  }
#endif

  // epilogue: C element e of lane l maps to (M = e + 8*half, N = l&15)
#pragma unroll
  for (int i = 0; i < 2; ++i) {
#pragma unroll
    for (int j = 0; j < 2; ++j) {
      const size_t mb = m0 + wm * 32 + i * 16 + half * 8;
      const size_t nn = n0 + wn * 32 + j * 16 + lm;
#pragma unroll
      for (int e = 0; e < 8; ++e) {
        const size_t idx = (mb + e) * (size_t)N + nn;
        if (MODE == 0) {
          const float g = acc1[i][j][e];
          const float u = acc2[i][j][e];
          const float zv = (g / (1.0f + __expf(-g))) * u;   // silu(g)*u
          out1bf[idx] = (__bf16)zv;
        } else {
          out1f[idx] = acc1[i][j][e];
          out2f[idx] = acc2[i][j][e];
        }
      }
    }
  }
}

// ---------------- k_seq = z @ lora_A^T (N = 16, WMMA from global) ----------------
__global__ __launch_bounds__(256)
void kseq_kernel(const __bf16* __restrict__ zbf, const __bf16* __restrict__ labf,
                 float* __restrict__ kseq) {
  const int tid  = threadIdx.x;
  const int w    = tid >> 5, l = tid & 31;
  const int half = l >> 4,   lm = l & 15;
  const size_t m0 = (size_t)blockIdx.x * 128 + (size_t)w * 16;

  const __bf16* arow = zbf + (m0 + lm) * (size_t)Ic;
  const __bf16* brow = labf + (size_t)lm * Ic;

  v8f acc;
#pragma unroll
  for (int e = 0; e < 8; ++e) acc[e] = 0.f;

  for (int k = 0; k < Ic; k += 32) {
    v16bf a = load_fragA(arow + k, half);
    v16bf b = load_fragB(brow + k, half);
    acc = wmma_bf16(a, b, acc);
  }
#pragma unroll
  for (int e = 0; e < 8; ++e)
    kseq[(m0 + half * 8 + e) * (size_t)Rc + lm] = acc[e];
}

// ---------------- delta-rule scan + LoRA + final add ----------------
// one lane per (b,h): dB[16] state in registers, k_t chunks staged in LDS.
__global__ __launch_bounds__(256)
void scan_kernel(const float* __restrict__ kseq, const float* __restrict__ vseq,
                 const float* __restrict__ loraB, float* __restrict__ out) {
  const int b = blockIdx.y;
  const int h = blockIdx.x * 256 + threadIdx.x;

  float dB[Rc], lB[Rc];
#pragma unroll
  for (int r = 0; r < Rc; ++r) {
    dB[r] = 0.f;
    lB[r] = loraB[(size_t)h * Rc + r];
  }

  __shared__ float kch[128 * Rc];
  const size_t bb = (size_t)b * Sc;

  for (int t0 = 0; t0 < Sc; t0 += 128) {
    const F4* src = (const F4*)(kseq + (bb + t0) * Rc);
    F4* dst = (F4*)kch;
    dst[threadIdx.x]       = src[threadIdx.x];
    dst[256 + threadIdx.x] = src[256 + threadIdx.x];
    __syncthreads();

    for (int tt = 0; tt < 128; ++tt) {
      const size_t gi = (bb + t0 + tt) * (size_t)Hc + h;
      const float v    = __builtin_nontemporal_load(vseq + gi);  // read-once stream
      const float base = out[gi];

      float kk[Rc];
#pragma unroll
      for (int q = 0; q < 4; ++q) {
        F4 f = *(const F4*)&kch[tt * Rc + q * 4];
        kk[q * 4 + 0] = f.x; kk[q * 4 + 1] = f.y;
        kk[q * 4 + 2] = f.z; kk[q * 4 + 3] = f.w;
      }

      float pred = 0.f, lora = 0.f;
#pragma unroll
      for (int r = 0; r < Rc; ++r) { pred += dB[r] * kk[r]; lora += lB[r] * kk[r]; }

      out[gi] = base + pred + lora;            // LORA_SCALING == 1
      const float e = TTT_LR * (pred - v);
#pragma unroll
      for (int r = 0; r < Rc; ++r) dB[r] -= e * kk[r];
    }
    __syncthreads();
  }
}

// ---------------- workspace layout ----------------
constexpr size_t OFF_XBF = 0;
constexpr size_t SZ_XBF  = (size_t)Mc * Hc * 2;
constexpr size_t OFF_WG  = OFF_XBF + SZ_XBF;
constexpr size_t SZ_W    = (size_t)Ic * Hc * 2;
constexpr size_t OFF_WU  = OFF_WG + SZ_W;
constexpr size_t OFF_WD  = OFF_WU + SZ_W;
constexpr size_t OFF_WT  = OFF_WD + SZ_W;
constexpr size_t OFF_LA  = OFF_WT + SZ_W;
constexpr size_t SZ_LA   = (size_t)Rc * Ic * 2;
constexpr size_t OFF_Z   = OFF_LA + SZ_LA;
constexpr size_t SZ_Z    = (size_t)Mc * Ic * 2;
constexpr size_t OFF_V   = OFF_Z + SZ_Z;
constexpr size_t SZ_V    = (size_t)Mc * Hc * 4;
constexpr size_t OFF_K   = OFF_V + SZ_V;

extern "C" void kernel_launch(void* const* d_in, const int* in_sizes, int n_in,
                              void* d_out, int out_size, void* d_ws, size_t ws_size,
                              hipStream_t stream) {
  (void)in_sizes; (void)n_in; (void)out_size; (void)ws_size;
  const float* x  = (const float*)d_in[0];
  const float* Wg = (const float*)d_in[1];
  const float* Wu = (const float*)d_in[2];
  const float* Wd = (const float*)d_in[3];
  const float* lA = (const float*)d_in[4];
  const float* lB = (const float*)d_in[5];
  const float* Wt = (const float*)d_in[6];
  float* out = (float*)d_out;

  char* ws = (char*)d_ws;
  __bf16* xbf  = (__bf16*)(ws + OFF_XBF);
  __bf16* wgbf = (__bf16*)(ws + OFF_WG);
  __bf16* wubf = (__bf16*)(ws + OFF_WU);
  __bf16* wdbf = (__bf16*)(ws + OFF_WD);
  __bf16* wtbf = (__bf16*)(ws + OFF_WT);
  __bf16* labf = (__bf16*)(ws + OFF_LA);
  __bf16* zbf  = (__bf16*)(ws + OFF_Z);
  float*  vseq = (float*)(ws + OFF_V);
  float*  kseq = (float*)(ws + OFF_K);

  // fp32 -> bf16 conversions (working set then mostly lives in the 192MB L2)
  {
    const int nx = Mc * Hc / 4, nw = Ic * Hc / 4, nl = Rc * Ic / 4;
    cvt_bf16_kernel<<<nx / 256, 256, 0, stream>>>(x,  xbf,  nx);
    cvt_bf16_kernel<<<nw / 256, 256, 0, stream>>>(Wg, wgbf, nw);
    cvt_bf16_kernel<<<nw / 256, 256, 0, stream>>>(Wu, wubf, nw);
    cvt_bf16_kernel<<<nw / 256, 256, 0, stream>>>(Wd, wdbf, nw);
    cvt_bf16_kernel<<<nw / 256, 256, 0, stream>>>(Wt, wtbf, nw);
    cvt_bf16_kernel<<<nl / 256, 256, 0, stream>>>(lA, labf, nl);
  }

  // z = silu(x Wg^T) * (x Wu^T)   [4096 x 8192], bf16
  gemm_dual_kernel<0><<<dim3(Ic / 128, Mc / 64), 256, 0, stream>>>(
      xbf, wgbf, wubf, Hc, Ic, nullptr, nullptr, zbf);

  // out_static(base) = z Wd^T -> d_out ; v_seq = z Wttt^T -> ws   [4096 x 2048] fp32
  gemm_dual_kernel<1><<<dim3(Hc / 128, Mc / 64), 256, 0, stream>>>(
      zbf, wdbf, wtbf, Ic, Hc, out, vseq, nullptr);

  // k_seq = z lora_A^T   [4096 x 16] fp32
  kseq_kernel<<<Mc / 128, 256, 0, stream>>>(zbf, labf, kseq);

  // delta-rule scan + LoRA term + final accumulation into d_out
  scan_kernel<<<dim3(Hc / 256, Bc), 256, 0, stream>>>(kseq, vseq, lB, out);
}
